// BidirectionalMambaBlock_35948876268409
// MI455X (gfx1250) — matrix operand order repaired and checked by reading, not compiled, not run on hardware
//
#include <hip/hip_runtime.h>

typedef _Float16 h16;
typedef __attribute__((ext_vector_type(16))) _Float16     v16h;
typedef __attribute__((ext_vector_type(8)))  float        v8f;
typedef __attribute__((ext_vector_type(4)))  unsigned int v4u;
typedef __attribute__((ext_vector_type(4)))  unsigned int u32x4;
typedef __attribute__((ext_vector_type(8)))  int          i32x8;
typedef __attribute__((ext_vector_type(4)))  int          i32x4;

union FragAB {
  v16h h;
  struct { v4u lo, hi; } q;
};

// ---------------------------------------------------------------------------
// Tensor Data Mover: 2D tile load (tile_h rows x tile_w elements, f16) from
// global memory into LDS. Descriptor built per CDNA5 ISA ch.8 (D# groups 0/1);
// trailing groups zero (2D tensor). data_size code 1 = 2 bytes.
// This toolchain exposes the 6-arg builtin:
//   (u32x4 g0, i32x8 g1, i32x4, i32x4, i32x8, i32 cpol)
// ---------------------------------------------------------------------------
__device__ static inline void tdm_load_2d_f16(unsigned lds_off, const void* gptr,
                                              unsigned tile_w, unsigned tile_h,
                                              unsigned stride_elems)
{
  const unsigned long long ga = (unsigned long long)(uintptr_t)gptr;
  u32x4 g0;
  g0[0] = 1u;                                            // count=1 (valid user D#)
  g0[1] = lds_off;                                       // lds_addr[31:0]
  g0[2] = (unsigned)(ga & 0xFFFFFFFFu);                  // global_addr[31:0]
  g0[3] = (unsigned)((ga >> 32) & 0x1FFFFFFu) | (2u << 30);  // addr[56:32], type=2
  i32x8 g1;
  g1[0] = (int)(1u << 16);                               // data_size=1 (2B), mask=0
  g1[1] = (int)((tile_w & 0xFFFFu) << 16);               // tensor_dim0[15:0]
  g1[2] = (int)(((tile_w >> 16) & 0xFFFFu) |             // tensor_dim0[31:16]
                ((tile_h & 0xFFFFu) << 16));             // tensor_dim1[15:0]
  g1[3] = (int)(((tile_h >> 16) & 0xFFFFu) |             // tensor_dim1[31:16]
                ((tile_w & 0xFFFFu) << 16));             // tile_dim0
  g1[4] = (int)(tile_h & 0xFFFFu);                       // tile_dim1 (tile_dim2=0)
  g1[5] = (int)stride_elems;                             // tensor_dim0_stride[31:0]
  g1[6] = 0;                                             // stride[47:32], dim1_stride lo
  g1[7] = 0;
  const i32x4 z4 = {0, 0, 0, 0};
  const i32x8 z8 = {0, 0, 0, 0, 0, 0, 0, 0};
  __builtin_amdgcn_tensor_load_to_lds(g0, g1, z4, z4, z8, 0);
}

// ---------------------------------------------------------------------------
// C[M,N] = A[M,K] * W[N,K]^T, A/W f16, C f32.
// Block = 128 threads (4 wave32 waves), block tile 64x64, wave tile 32x32
// (2x2 register blocking, 4 WMMA accumulators). K tiled by 32 through LDS:
// W tile arrives via the Tensor Data Mover (async), A tile via b128 loads.
// M % 64 == 0, N % 64 == 0, K % 32 == 0 for every call here.
// ---------------------------------------------------------------------------
__global__ __launch_bounds__(128) void wmma_gemm(
    const h16* __restrict__ A, int lda,
    const h16* __restrict__ W, int ldw,
    float* __restrict__ C, int ldc, int K)
{
  __shared__ __align__(16) h16 As[64 * 32];
  __shared__ __align__(16) h16 Ws[64 * 32];
  const int tid  = threadIdx.x;
  const int wave = tid >> 5;
  const int lane = tid & 31;
  const int l15  = lane & 15;
  const size_t bm = (size_t)blockIdx.y * 64;
  const size_t bn = (size_t)blockIdx.x * 64;
  const int wm = (wave >> 1) * 32;
  const int wn = (wave & 1) * 32;
  const bool issuer = (__builtin_amdgcn_readfirstlane(wave) == 0);

  const int r0 = tid >> 2;            // A-tile chunk row (0..31); +32 for chunk 1
  const int q0 = (tid & 3) * 8;       // 16B chunk offset within row (halves)

  v8f acc00 = {}, acc01 = {}, acc10 = {}, acc11 = {};

  for (int k0 = 0; k0 < K; k0 += 32) {
    // --- W tile (64 x 32 halves) via TDM: one DMA per block per K-step -----
    if (issuer)
      tdm_load_2d_f16((unsigned)(uintptr_t)&Ws[0], W + bn * (size_t)ldw + k0,
                      32u, 64u, (unsigned)ldw);
    // --- A tile: each thread moves two b128 chunks --------------------------
    {
      const v4u va0 = *(const v4u*)(A + (bm + r0) * (size_t)lda + k0 + q0);
      const v4u va1 = *(const v4u*)(A + (bm + 32 + r0) * (size_t)lda + k0 + q0);
      *(v4u*)&As[r0 * 32 + q0]        = va0;
      *(v4u*)&As[(32 + r0) * 32 + q0] = va1;
    }
    if (k0 + 32 < K)
      __builtin_prefetch(A + (bm + r0) * (size_t)lda + k0 + 32, 0, 0);
    if (issuer) __builtin_amdgcn_s_wait_tensorcnt(0);
    __syncthreads();

    // --- fragments ----------------------------------------------------------
    // A (16-bit 16x32): lane<16 -> M=l15, K={0..7,16..23}; lane>=16 -> K={8..15,24..31}
    // B (16-bit 32x16): lane = N column; lanes 0-15 K={0..7,16..23}, lanes 16-31 K={8..15,24..31}
    FragAB a0, a1, b0, b1;
    const int kbA = (lane < 16) ? 0 : 8;
    const int kbB = (lane < 16) ? 0 : 8;
    const h16* ar0 = As + (wm + l15) * 32 + kbA;
    const h16* ar1 = As + (wm + 16 + l15) * 32 + kbA;
    a0.q.lo = *(const v4u*)ar0;  a0.q.hi = *(const v4u*)(ar0 + 16);
    a1.q.lo = *(const v4u*)ar1;  a1.q.hi = *(const v4u*)(ar1 + 16);
    const h16* br0 = Ws + (wn + l15) * 32 + kbB;
    const h16* br1 = Ws + (wn + 16 + l15) * 32 + kbB;
    b0.q.lo = *(const v4u*)br0;  b0.q.hi = *(const v4u*)(br0 + 16);
    b1.q.lo = *(const v4u*)br1;  b1.q.hi = *(const v4u*)(br1 + 16);

    acc00 = __builtin_amdgcn_wmma_f32_16x16x32_f16(false, a0.h, false, b0.h,
                                                   (short)0, acc00, false, false);
    acc01 = __builtin_amdgcn_wmma_f32_16x16x32_f16(false, a0.h, false, b1.h,
                                                   (short)0, acc01, false, false);
    acc10 = __builtin_amdgcn_wmma_f32_16x16x32_f16(false, a1.h, false, b0.h,
                                                   (short)0, acc10, false, false);
    acc11 = __builtin_amdgcn_wmma_f32_16x16x32_f16(false, a1.h, false, b1.h,
                                                   (short)0, acc11, false, false);
    __syncthreads();
  }

  // C/D layout: VGPR r -> M = r (+8 for lanes 16-31), N = lane&15
  const size_t n0 = bn + wn + l15;
  const size_t m0 = bm + wm + ((lane < 16) ? 0 : 8);
#pragma unroll
  for (int r = 0; r < 8; ++r) {
    C[(m0 + r) * (size_t)ldc + n0]           = acc00[r];
    C[(m0 + r) * (size_t)ldc + n0 + 16]      = acc01[r];
    C[(m0 + 16 + r) * (size_t)ldc + n0]      = acc10[r];
    C[(m0 + 16 + r) * (size_t)ldc + n0 + 16] = acc11[r];
  }
}

// ---------------------------------------------------------------------------
// LayerNorm -> f16 (normal order + time-flipped copy for the bwd direction)
// ---------------------------------------------------------------------------
__global__ __launch_bounds__(256) void layernorm_kernel(
    const float* __restrict__ x, const float* __restrict__ g,
    const float* __restrict__ beta, h16* __restrict__ xnh,
    h16* __restrict__ xnflip, int L, int D)
{
  const int row = blockIdx.x;      // b*L + l
  const int b = row / L, l = row % L;
  const float* xr = x + (size_t)row * D;
  __shared__ float ss[256], sq[256];
  float s = 0.f, s2 = 0.f;
  for (int i = threadIdx.x; i < D; i += 256) { float v = xr[i]; s += v; s2 += v * v; }
  ss[threadIdx.x] = s; sq[threadIdx.x] = s2;
  __syncthreads();
  for (int o = 128; o > 0; o >>= 1) {
    if (threadIdx.x < o) { ss[threadIdx.x] += ss[threadIdx.x + o];
                           sq[threadIdx.x] += sq[threadIdx.x + o]; }
    __syncthreads();
  }
  const float mean = ss[0] / D;
  const float var  = sq[0] / D - mean * mean;
  const float rstd = rsqrtf(var + 1e-5f);
  h16* outr  = xnh    + (size_t)row * D;
  h16* outfr = xnflip + ((size_t)b * L + (L - 1 - l)) * D;
  for (int i = threadIdx.x; i < D; i += 256) {
    const float v = (xr[i] - mean) * rstd * g[i] + beta[i];
    const h16 hv = (h16)v;
    outr[i] = hv;
    outfr[i] = hv;
  }
}

__global__ void cvt_f32_f16(const float* __restrict__ s, h16* __restrict__ d, size_t n)
{
  const size_t i = (size_t)blockIdx.x * 256 + threadIdx.x;
  if (i < n) d[i] = (h16)s[i];
}

// f32[src_rows, cols] -> f16[dst_rows, cols], zero-padding rows >= src_rows
__global__ void cvt_pad_f32_f16(const float* __restrict__ s, h16* __restrict__ d,
                                int src_rows, int cols, size_t n)
{
  const size_t i = (size_t)blockIdx.x * 256 + threadIdx.x;
  if (i < n) {
    const int r = (int)(i / cols);
    const int c = (int)(i % cols);
    d[i] = (r < src_rows) ? (h16)s[(size_t)r * cols + c] : (h16)0.f;
  }
}

// ---------------------------------------------------------------------------
// Causal depthwise conv (width 4) over the xi half of xz, + bias + SiLU.
// ---------------------------------------------------------------------------
__global__ __launch_bounds__(256) void conv_silu_kernel(
    const float* __restrict__ xz, const float* __restrict__ cw,
    const float* __restrict__ cb, float* __restrict__ xi,
    h16* __restrict__ xih, int L)
{
  const int DI = 2048;
  const size_t idx = (size_t)blockIdx.x * 256 + threadIdx.x;  // B*L*DI
  const int d = (int)(idx & (DI - 1));
  const size_t bl = idx >> 11;
  const int l = (int)(bl % L);
  const int b = (int)(bl / L);
  float acc = cb[d];
#pragma unroll
  for (int j = 0; j < 4; ++j) {
    const int ls = l - 3 + j;
    if (ls >= 0)
      acc += cw[d * 4 + j] * xz[((size_t)b * L + ls) * 4096 + d];
  }
  const float sv = acc / (1.f + __expf(-acc));   // SiLU
  xi[idx]  = sv;
  xih[idx] = (h16)sv;
}

// ---------------------------------------------------------------------------
// Selective scan: one thread per (b, d) channel, h[16] state in VGPRs,
// per-timestep B/C broadcast through LDS. proj row stride = 128 (padded).
// ---------------------------------------------------------------------------
__global__ __launch_bounds__(256) void scan_kernel(
    const float* __restrict__ dpre, const float* __restrict__ dt_b,
    const float* __restrict__ xi, const float* __restrict__ xz,
    const float* __restrict__ proj, const float* __restrict__ A_log,
    const float* __restrict__ Dp, h16* __restrict__ ygate, int L)
{
  const int DI = 2048;
  const int b = blockIdx.y;
  const int d = blockIdx.x * 256 + threadIdx.x;
  float A[16];
#pragma unroll
  for (int s = 0; s < 16; ++s) A[s] = -__expf(A_log[d * 16 + s]);
  const float dtb = dt_b[d];
  const float Dd  = Dp[d];
  float h[16];
#pragma unroll
  for (int s = 0; s < 16; ++s) h[s] = 0.f;
  __shared__ float BC[32];
  for (int t = 0; t < L; ++t) {
    const size_t row = (size_t)b * L + t;
    if (threadIdx.x < 32) BC[threadIdx.x] = proj[row * 128 + 64 + threadIdx.x];
    __syncthreads();
    const float dp    = dpre[row * DI + d] + dtb;
    const float delta = dp > 20.f ? dp : log1pf(__expf(dp));  // softplus
    const float xv    = xi[row * DI + d];
    const float zv    = xz[row * (2 * DI) + DI + d];
    const float dx    = delta * xv;
    float y = 0.f;
#pragma unroll
    for (int s = 0; s < 16; ++s) {
      h[s] = __expf(delta * A[s]) * h[s] + dx * BC[s];
      y = fmaf(h[s], BC[16 + s], y);
    }
    y = fmaf(xv, Dd, y);
    const float gz = zv / (1.f + __expf(-zv));
    ygate[row * DI + d] = (h16)(y * gz);
    __syncthreads();
  }
}

// Pack per-direction out_proj result into the concat buffer (un-flip for bwd)
__global__ void pack_cat_kernel(const float* __restrict__ yout,
                                h16* __restrict__ ycat, int L, int flip)
{
  const size_t idx = (size_t)blockIdx.x * 256 + threadIdx.x;  // B*L*1024
  const int d = (int)(idx & 1023);
  const size_t bl = idx >> 10;
  const int l = (int)(bl % L);
  const int b = (int)(bl / L);
  const size_t drow = flip ? ((size_t)b * L + (L - 1 - l)) : bl;
  ycat[drow * 2048 + (flip ? 1024 : 0) + d] = (h16)yout[idx];
}

__global__ void final_kernel(const float* __restrict__ x, const float* __restrict__ t,
                             const float* __restrict__ mb, float* __restrict__ out,
                             size_t n)
{
  const size_t i = (size_t)blockIdx.x * 256 + threadIdx.x;
  if (i < n) out[i] = x[i] + t[i] + mb[i & 1023];
}

// ---------------------------------------------------------------------------
extern "C" void kernel_launch(void* const* d_in, const int* in_sizes, int n_in,
                              void* d_out, int out_size, void* d_ws, size_t ws_size,
                              hipStream_t stream)
{
  (void)in_sizes; (void)n_in; (void)out_size; (void)ws_size;
  const int B = 4, L = 2048, DM = 1024, DI = 2048, DTR = 64;
  const size_t BL = (size_t)B * L;

  const float* x       = (const float*)d_in[0];
  const float* ln_g    = (const float*)d_in[1];
  const float* ln_b    = (const float*)d_in[2];
  const float* merge_w = (const float*)d_in[3];
  const float* merge_b = (const float*)d_in[4];
  struct Dir { const float *in_w, *conv_w, *conv_b, *xproj_w, *dt_w, *dt_b,
                           *A_log, *D, *out_w; };
  Dir dirs[2];
  for (int j = 0; j < 2; ++j) {
    const int o = 5 + j * 9;
    dirs[j] = { (const float*)d_in[o + 0], (const float*)d_in[o + 1],
                (const float*)d_in[o + 2], (const float*)d_in[o + 3],
                (const float*)d_in[o + 4], (const float*)d_in[o + 5],
                (const float*)d_in[o + 6], (const float*)d_in[o + 7],
                (const float*)d_in[o + 8] };
  }

  char* p = (char*)d_ws;
  auto alloc = [&](size_t bytes) -> char* {
    char* r = p; p += (bytes + 255) & ~(size_t)255; return r;
  };
  h16*   xn16       = (h16*)alloc(BL * DM * 2);
  h16*   xnflip16   = (h16*)alloc(BL * DM * 2);
  h16*   inw16[2]   = { (h16*)alloc((size_t)2 * DI * DM * 2),
                        (h16*)alloc((size_t)2 * DI * DM * 2) };
  h16*   xprojw16[2]= { (h16*)alloc((size_t)128 * DI * 2),     // padded 96 -> 128
                        (h16*)alloc((size_t)128 * DI * 2) };
  h16*   dtw16[2]   = { (h16*)alloc((size_t)DI * DTR * 2),
                        (h16*)alloc((size_t)DI * DTR * 2) };
  h16*   outw16[2]  = { (h16*)alloc((size_t)DM * DI * 2),
                        (h16*)alloc((size_t)DM * DI * 2) };
  h16*   mergew16   = (h16*)alloc((size_t)DM * 2 * DM * 2);
  float* xz         = (float*)alloc(BL * 2 * DI * 4);
  float* xi         = (float*)alloc(BL * DI * 4);
  h16*   xi16       = (h16*)alloc(BL * DI * 2);
  float* proj       = (float*)alloc(BL * 128 * 4);             // padded row stride
  h16*   proj16     = (h16*)alloc(BL * 128 * 2);
  float* dpre       = (float*)alloc(BL * DI * 4);
  h16*   ygate16    = (h16*)alloc(BL * DI * 2);
  float* yout       = (float*)alloc(BL * DM * 4);
  h16*   ycat16     = (h16*)alloc(BL * 2 * DM * 2);
  float* tmp        = (float*)alloc(BL * DM * 4);

  auto g1 = [](size_t n) { return dim3((unsigned)((n + 255) / 256)); };

  layernorm_kernel<<<dim3((unsigned)BL), 256, 0, stream>>>(x, ln_g, ln_b,
                                                           xn16, xnflip16, L, DM);
  cvt_f32_f16<<<g1((size_t)DM * 2 * DM), 256, 0, stream>>>(
      merge_w, mergew16, (size_t)DM * 2 * DM);
  for (int j = 0; j < 2; ++j) {
    cvt_f32_f16<<<g1((size_t)2 * DI * DM), 256, 0, stream>>>(
        dirs[j].in_w, inw16[j], (size_t)2 * DI * DM);
    cvt_pad_f32_f16<<<g1((size_t)128 * DI), 256, 0, stream>>>(
        dirs[j].xproj_w, xprojw16[j], 96, DI, (size_t)128 * DI);
    cvt_f32_f16<<<g1((size_t)DI * DTR), 256, 0, stream>>>(
        dirs[j].dt_w, dtw16[j], (size_t)DI * DTR);
    cvt_f32_f16<<<g1((size_t)DM * DI), 256, 0, stream>>>(
        dirs[j].out_w, outw16[j], (size_t)DM * DI);
  }

  const unsigned MT = (unsigned)(BL / 64);  // 128 M-tiles
  for (int j = 0; j < 2; ++j) {
    const h16* Ain = (j == 0) ? xn16 : xnflip16;
    // xz = xn @ in_w^T                       [8192 x 4096], K=1024
    wmma_gemm<<<dim3(4096 / 64, MT), 128, 0, stream>>>(Ain, DM, inw16[j], DM,
                                                       xz, 4096, DM);
    conv_silu_kernel<<<g1(BL * DI), 256, 0, stream>>>(xz, dirs[j].conv_w,
                                                      dirs[j].conv_b, xi, xi16, L);
    // proj = xi @ xproj_w^T (N padded 96->128)  [8192 x 128], K=2048
    wmma_gemm<<<dim3(128 / 64, MT), 128, 0, stream>>>(xi16, DI, xprojw16[j], DI,
                                                      proj, 128, DI);
    cvt_f32_f16<<<g1(BL * 128), 256, 0, stream>>>(proj, proj16, BL * 128);
    // dpre = dt @ dt_w^T  (dt = proj[:, :64], lda=128)  [8192 x 2048], K=64
    wmma_gemm<<<dim3(DI / 64, MT), 128, 0, stream>>>(proj16, 128, dtw16[j], DTR,
                                                     dpre, DI, DTR);
    scan_kernel<<<dim3(DI / 256, B), 256, 0, stream>>>(
        dpre, dirs[j].dt_b, xi, xz, proj, dirs[j].A_log, dirs[j].D, ygate16, L);
    // yout = ygate @ out_w^T                 [8192 x 1024], K=2048
    wmma_gemm<<<dim3(DM / 64, MT), 128, 0, stream>>>(ygate16, DI, outw16[j], DI,
                                                     yout, DM, DI);
    pack_cat_kernel<<<g1(BL * DM), 256, 0, stream>>>(yout, ycat16, L, j);
  }
  // tmp = concat(y_f, y_b) @ merge_w^T       [8192 x 1024], K=2048
  wmma_gemm<<<dim3(DM / 64, MT), 128, 0, stream>>>(ycat16, 2 * DM, mergew16,
                                                   2 * DM, tmp, DM, 2 * DM);
  final_kernel<<<g1(BL * DM), 256, 0, stream>>>(x, tmp, merge_b, (float*)d_out,
                                                BL * DM);
}
